// RGCN_42846593744973
// MI455X (gfx1250) — matrix operand order repaired and checked by reading, not compiled, not run on hardware
//
#include <hip/hip_runtime.h>

#define N_NODES 100000
#define N_EDGES 1600000
#define INP     16
#define EMB     32
#define ATTN    32
#define NBASES  4
#define D0      48      // INP + EMB
#define K0PAD   64      // D0 padded to multiple of 32
#define NC_L    160     // 4*32 basis cols + 32 self-loop cols
#define NC_ATT  64      // 32 a1 cols + 32 a2 cols

typedef __attribute__((ext_vector_type(16))) _Float16 v16h;
typedef __attribute__((ext_vector_type(8)))  _Float16 v8h;
typedef __attribute__((ext_vector_type(8)))  float    v8f;

// ---------------------------------------------------------------------------
// Prep: h0 = [feat | embed[node_ids]] padded to K0PAD, stored f16.
// ---------------------------------------------------------------------------
__global__ void prep_h0(const float* __restrict__ feat, const int* __restrict__ node_ids,
                        const float* __restrict__ embed, _Float16* __restrict__ h0) {
    int idx = blockIdx.x * blockDim.x + threadIdx.x;
    if (idx >= N_NODES * K0PAD) return;
    int n = idx >> 6, k = idx & 63;
    float v = 0.0f;
    if (k < INP)      v = feat[n * INP + k];
    else if (k < D0)  v = embed[(size_t)node_ids[n] * EMB + (k - INP)];
    h0[idx] = (_Float16)v;
}

// Pack attention weights: Wt[c][k] (c<32: A1_w[c,k], else A2_w[c-32,k]), K padded to 64.
__global__ void pack_att(const float* __restrict__ A1_w, const float* __restrict__ A2_w,
                         _Float16* __restrict__ Wt) {
    int idx = blockIdx.x * blockDim.x + threadIdx.x;
    if (idx >= NC_ATT * K0PAD) return;
    int c = idx >> 6, k = idx & 63;
    float v = 0.0f;
    if (k < D0) v = (c < ATTN) ? A1_w[c * D0 + k] : A2_w[(c - ATTN) * D0 + k];
    Wt[idx] = (_Float16)v;
}

// Pack layer weights transposed: Wt[c][k]; c<128 -> bases[c/32][k][c%32], else sloop[k][c-128].
__global__ void pack_layer(const float* __restrict__ bases, const float* __restrict__ sloop,
                           _Float16* __restrict__ Wt, int di, int kpad) {
    int idx = blockIdx.x * blockDim.x + threadIdx.x;
    if (idx >= NC_L * kpad) return;
    int c = idx / kpad, k = idx % kpad;
    float v = 0.0f;
    if (k < di)
        v = (c < 128) ? bases[((size_t)(c >> 5) * di + k) * 32 + (c & 31)]
                      : sloop[(size_t)k * 32 + (c - 128)];
    Wt[idx] = (_Float16)v;
}

__global__ void zero_f32(float* __restrict__ p, int n) {
    int i = blockIdx.x * blockDim.x + threadIdx.x;
    if (i < n) p[i] = 0.0f;
}

// ---------------------------------------------------------------------------
// WMMA GEMM: out[N_NODES x NC] = A[N_NODES x KPAD](f16) * Wt^T, Wt is [NC x KPAD] f16.
// One wave per 16-row strip; loops over NC/16 column tiles, KPAD/32 K tiles.
// Fragment layouts (wave32): A and B both use contiguous K, k = 16*hi + s; D as in the ISA.
// ---------------------------------------------------------------------------
template <int KPAD, int NC, typename OutT>
__global__ __launch_bounds__(256)
void wmma_gemm(const _Float16* __restrict__ A, const _Float16* __restrict__ Wt,
               OutT* __restrict__ out) {
    const int lane = threadIdx.x & 31;
    const int wave = (blockIdx.x * blockDim.x + threadIdx.x) >> 5;
    const int m0 = wave * 16;
    if (m0 >= N_NODES) return;                 // wave-uniform: EXEC stays all-ones

    const int hi  = lane >> 4;                 // 0: lanes 0-15, 1: lanes 16-31
    const int l15 = lane & 15;
    constexpr int KT = KPAD / 32;

    // A fragments: row M = m0 + (lane&15); 16 contiguous K values starting at 16*hi.
    v16h a[KT];
    const _Float16* arow = A + (size_t)(m0 + l15) * KPAD;
#pragma unroll
    for (int kt = 0; kt < KT; ++kt) {
        const _Float16* base = arow + kt * 32;
        v8h lo = *(const v8h*)(base + (hi ? 16 : 0));
        v8h hh = *(const v8h*)(base + (hi ? 24 : 8));
#pragma unroll
        for (int i = 0; i < 8; ++i) { a[kt][i] = lo[i]; a[kt][i + 8] = hh[i]; }
    }

#pragma unroll
    for (int ct = 0; ct < NC / 16; ++ct) {
        v8f acc = {};
#pragma unroll
        for (int kt = 0; kt < KT; ++kt) {
            // B fragment: lane holds col N = ct*16 + (lane&15), 16 contiguous K values.
            const v16h b = *(const v16h*)(Wt + (size_t)(ct * 16 + l15) * KPAD + kt * 32 + 16 * hi);
            acc = __builtin_amdgcn_wmma_f32_16x16x32_f16(
                /*neg_a=*/false, a[kt], /*neg_b=*/false, b,
                /*c_mod=*/(short)0, acc, /*reuse_a=*/false, /*reuse_b=*/false);
        }
        // D: VGPR r -> row M = r + 8*hi, col N = lane&15
#pragma unroll
        for (int r = 0; r < 8; ++r)
            out[(size_t)(m0 + hi * 8 + r) * NC + ct * 16 + l15] = (OutT)acc[r];
    }
}

// ---------------------------------------------------------------------------
// Edge alpha: wave per edge, lane = attention dim.
// alpha = sigmoid(GAMMA - sqrt(||a1[src] + rel[type] - a2[dst]||^2 + 1e-12))
// ---------------------------------------------------------------------------
__global__ __launch_bounds__(256)
void edge_alpha(const int* __restrict__ src, const int* __restrict__ dst,
                const int* __restrict__ typ, const float* __restrict__ att,
                const float* __restrict__ attn_rel,
                const float* __restrict__ a1b, const float* __restrict__ a2b,
                float* __restrict__ alpha) {
    int e = (blockIdx.x * blockDim.x + threadIdx.x) >> 5;
    if (e >= N_EDGES) return;
    int lane = threadIdx.x & 31;
    int s = src[e], d = dst[e], t = typ[e];
    float diff = att[(size_t)s * NC_ATT + lane] + a1b[lane]
               + attn_rel[(size_t)t * ATTN + lane]
               - att[(size_t)d * NC_ATT + 32 + lane] - a2b[lane];
    float v = diff * diff;
#pragma unroll
    for (int m = 16; m >= 1; m >>= 1) v += __shfl_xor(v, m, 32);
    if (lane == 0)
        alpha[e] = 1.0f / (1.0f + __expf(sqrtf(v + 1e-12f) - 10.0f));
}

// ---------------------------------------------------------------------------
// Edge scatter: wave per edge, lane = output channel.
// msg = sum_b wcomp[type,b] * hb[src, b, lane];  agg[dst, lane] += alpha * msg
// ---------------------------------------------------------------------------
__global__ __launch_bounds__(256)
void edge_scatter(const int* __restrict__ src, const int* __restrict__ dst,
                  const int* __restrict__ typ, const float* __restrict__ alpha,
                  const float* __restrict__ wcomp, const _Float16* __restrict__ hb,
                  float* __restrict__ agg) {
    int e = (blockIdx.x * blockDim.x + threadIdx.x) >> 5;
    if (e >= N_EDGES) return;
    int lane = threadIdx.x & 31;
    int s = src[e], d = dst[e], t = typ[e];
    float al = alpha[e];
    const _Float16* row = hb + (size_t)s * NC_L;
    const float* wc = wcomp + (size_t)t * NBASES;
    float m = wc[0] * (float)row[lane]      + wc[1] * (float)row[32 + lane]
            + wc[2] * (float)row[64 + lane] + wc[3] * (float)row[96 + lane];
    atomicAdd(&agg[(size_t)d * EMB + lane], al * m);
}

// ---------------------------------------------------------------------------
// ReLU combine: h_new = relu(selfloop + agg + bias). Non-final -> f16 for next GEMM.
// ---------------------------------------------------------------------------
template <bool FINAL>
__global__ void relu_combine(const _Float16* __restrict__ hb, const float* __restrict__ agg,
                             const float* __restrict__ bias,
                             float* __restrict__ out_f32, _Float16* __restrict__ out_f16) {
    int idx = blockIdx.x * blockDim.x + threadIdx.x;
    if (idx >= N_NODES * EMB) return;
    int n = idx >> 5, o = idx & 31;
    float v = (float)hb[(size_t)n * NC_L + 128 + o] + agg[idx] + bias[o];
    v = v > 0.0f ? v : 0.0f;
    if (FINAL) out_f32[idx] = v;
    else       out_f16[idx] = (_Float16)v;
}

// ---------------------------------------------------------------------------
extern "C" void kernel_launch(void* const* d_in, const int* in_sizes, int n_in,
                              void* d_out, int out_size, void* d_ws, size_t ws_size,
                              hipStream_t stream) {
    const float* feat      = (const float*)d_in[0];
    const int*   node_ids  = (const int*)  d_in[1];
    const int*   edge_src  = (const int*)  d_in[2];
    const int*   edge_dst  = (const int*)  d_in[3];
    const int*   edge_type = (const int*)  d_in[4];
    const float* embed     = (const float*)d_in[5];
    const float* attn_rel  = (const float*)d_in[6];
    const float* A1_w = (const float*)d_in[7];
    const float* A1_b = (const float*)d_in[8];
    const float* A2_w = (const float*)d_in[9];
    const float* A2_b = (const float*)d_in[10];
    const float* bases[3] = { (const float*)d_in[11], (const float*)d_in[15], (const float*)d_in[19] };
    const float* wcomp[3] = { (const float*)d_in[12], (const float*)d_in[16], (const float*)d_in[20] };
    const float* sloop[3] = { (const float*)d_in[13], (const float*)d_in[17], (const float*)d_in[21] };
    const float* bias[3]  = { (const float*)d_in[14], (const float*)d_in[18], (const float*)d_in[22] };
    float* out = (float*)d_out;

    // Workspace carve-out (256B aligned)
    char* w = (char*)d_ws;
    size_t off = 0;
    auto carve = [&](size_t bytes) -> void* {
        void* p = w + off;
        off = (off + bytes + 255) & ~(size_t)255;
        return p;
    };
    _Float16* h0    = (_Float16*)carve((size_t)N_NODES * K0PAD * 2);
    _Float16* hcur  = (_Float16*)carve((size_t)N_NODES * EMB * 2);
    float*    att   = (float*)   carve((size_t)N_NODES * NC_ATT * 4);
    float*    alpha = (float*)   carve((size_t)N_EDGES * 4);
    _Float16* hb    = (_Float16*)carve((size_t)N_NODES * NC_L * 2);
    float*    agg   = (float*)   carve((size_t)N_NODES * EMB * 4);
    _Float16* Watt  = (_Float16*)carve((size_t)NC_ATT * K0PAD * 2);
    _Float16* W0    = (_Float16*)carve((size_t)NC_L * K0PAD * 2);
    _Float16* W1    = (_Float16*)carve((size_t)NC_L * EMB * 2);
    _Float16* W2    = (_Float16*)carve((size_t)NC_L * EMB * 2);

    const int TB = 256;
    const int gemmBlocks = (N_NODES / 16 + 7) / 8;       // 8 waves/block, 16 rows/wave
    const int edgeBlocks = N_EDGES / 8;                  // 8 waves/block, 1 edge/wave
    const int nodeElemBlocks = (N_NODES * EMB + TB - 1) / TB;

    // Stage 0: input/weight prep
    prep_h0<<<(N_NODES * K0PAD + TB - 1) / TB, TB, 0, stream>>>(feat, node_ids, embed, h0);
    pack_att<<<(NC_ATT * K0PAD + TB - 1) / TB, TB, 0, stream>>>(A1_w, A2_w, Watt);
    pack_layer<<<(NC_L * K0PAD + TB - 1) / TB, TB, 0, stream>>>(bases[0], sloop[0], W0, D0, K0PAD);
    pack_layer<<<(NC_L * EMB + TB - 1) / TB, TB, 0, stream>>>(bases[1], sloop[1], W1, EMB, EMB);
    pack_layer<<<(NC_L * EMB + TB - 1) / TB, TB, 0, stream>>>(bases[2], sloop[2], W2, EMB, EMB);

    // Stage 1: shared attention (a1|a2 projections + per-edge alpha)
    wmma_gemm<K0PAD, NC_ATT, float><<<gemmBlocks, TB, 0, stream>>>(h0, Watt, att);
    edge_alpha<<<edgeBlocks, TB, 0, stream>>>(edge_src, edge_dst, edge_type,
                                              att, attn_rel, A1_b, A2_b, alpha);

    // Layer 0 (K=64 padded input)
    wmma_gemm<K0PAD, NC_L, _Float16><<<gemmBlocks, TB, 0, stream>>>(h0, W0, hb);
    zero_f32<<<nodeElemBlocks, TB, 0, stream>>>(agg, N_NODES * EMB);
    edge_scatter<<<edgeBlocks, TB, 0, stream>>>(edge_src, edge_dst, edge_type,
                                                alpha, wcomp[0], hb, agg);
    relu_combine<false><<<nodeElemBlocks, TB, 0, stream>>>(hb, agg, bias[0], nullptr, hcur);

    // Layer 1 (K=32)
    wmma_gemm<EMB, NC_L, _Float16><<<gemmBlocks, TB, 0, stream>>>(hcur, W1, hb);
    zero_f32<<<nodeElemBlocks, TB, 0, stream>>>(agg, N_NODES * EMB);
    edge_scatter<<<edgeBlocks, TB, 0, stream>>>(edge_src, edge_dst, edge_type,
                                                alpha, wcomp[1], hb, agg);
    relu_combine<false><<<nodeElemBlocks, TB, 0, stream>>>(hb, agg, bias[1], nullptr, hcur);

    // Layer 2 (K=32) -> final f32 output
    wmma_gemm<EMB, NC_L, _Float16><<<gemmBlocks, TB, 0, stream>>>(hcur, W2, hb);
    zero_f32<<<nodeElemBlocks, TB, 0, stream>>>(agg, N_NODES * EMB);
    edge_scatter<<<edgeBlocks, TB, 0, stream>>>(edge_src, edge_dst, edge_type,
                                                alpha, wcomp[2], hb, agg);
    relu_combine<true><<<nodeElemBlocks, TB, 0, stream>>>(hb, agg, bias[2], out, nullptr);
}
